// TransformerLayer_44753559224898
// MI455X (gfx1250) — matrix operand order repaired and checked by reading, not compiled, not run on hardware
//
#include <hip/hip_runtime.h>
#include <hip/hip_bf16.h>

// ---------------------------------------------------------------- constants
#define BB   2
#define SS   2048
#define DD   1024
#define HH   16
#define DHH  64
#define CC   64
#define MM   (BB*SS)      // 4096 rows
#define DFF  (4*DD)       // 4096

#define FMINF (-3.4028234663852886e+38f)

typedef __attribute__((ext_vector_type(16))) __bf16 v16bf;
typedef __attribute__((ext_vector_type(8)))  __bf16 v8bf;
typedef __attribute__((ext_vector_type(8)))  float  v8f;
typedef unsigned int u32;
typedef __attribute__((ext_vector_type(4))) u32 v4u;
typedef __attribute__((ext_vector_type(8))) int v8i;
typedef __attribute__((ext_vector_type(4))) int v4i;

#ifndef __has_builtin
#define __has_builtin(x) 0
#endif
#if __has_builtin(__builtin_amdgcn_tensor_load_to_lds) && \
    __has_builtin(__builtin_amdgcn_s_wait_tensorcnt)
#define USE_TDM 1
#else
#define USE_TDM 0
#endif

static __device__ __forceinline__ __bf16 f2bf(float f) { return (__bf16)f; }

// A-fragment (16x32 bf16, row-major source row): elements 0..7 -> K = half*8+i,
// elements 8..15 -> K = 16+half*8+i   (ISA 7.12.2, 16-bit A 16x32)
static __device__ __forceinline__ v16bf frag_a_ld(const __bf16* rowBase, int half) {
    v8bf lo = *(const v8bf*)(rowBase + half * 8);
    v8bf hi = *(const v8bf*)(rowBase + 16 + half * 8);
    v16bf r;
#pragma unroll
    for (int i = 0; i < 8; ++i) { r[i] = lo[i]; r[i + 8] = hi[i]; }
    return r;
}

// B-fragment (32x16 bf16) from an [N][K] row: lane holds column n = lane&15,
// elements 0..7 -> K = half*8+i, elements 8..15 -> K = 16+half*8+i
static __device__ __forceinline__ v16bf frag_b_ld(const __bf16* rowBase, int half) {
    v8bf lo = *(const v8bf*)(rowBase + half * 8);
    v8bf hi = *(const v8bf*)(rowBase + 16 + half * 8);
    v16bf r;
#pragma unroll
    for (int i = 0; i < 8; ++i) { r[i] = lo[i]; r[i + 8] = hi[i]; }
    return r;
}

static __device__ __forceinline__ v8f wmma_bf16(v16bf a, v16bf b, v8f c) {
    return __builtin_amdgcn_wmma_f32_16x16x32_bf16(false, a, false, b, (short)0, c,
                                                   false, false);
}

#if USE_TDM
// --------------------------------------------------------- TDM 2D tile load
// DMA a (tileRows x 32) bf16 tile (row stride = Kdim elements) into LDS.
// Hardware padding: pad_interval=16 DWORDs (one 64B row), pad_amount=4 DWORDs
// (16B) -> LDS row stride = 40 bf16 elements, matching the fragment loader.
static __device__ __forceinline__ void tdm_load_tile(u32 lds_off,
                                                     const __bf16* gptr,
                                                     u32 Kdim, u32 tensorRows,
                                                     u32 tileRows) {
    unsigned long long ga = (unsigned long long)(__SIZE_TYPE__)gptr;
    v4u g0 = {1u,                                    // count=1 (valid user D#)
              lds_off,                               // lds_addr (bytes)
              (u32)ga,                               // global_addr[31:0]
              (u32)((ga >> 32) & 0x01FFFFFFu) | (2u << 30)}; // addr[56:32]|type=2
    v8i g1;
    g1[0] = (int)((1u << 16)      // data_size = 2 bytes
                | (1u << 20)      // pad_enable
                | (3u << 22)      // pad_interval: 16 DWORDs
                | (3u << 25));    // pad_amount:   4 DWORDs
    g1[1] = (int)((Kdim & 0xFFFFu) << 16);                       // tensor_dim0 lo
    g1[2] = (int)(((Kdim >> 16) & 0xFFFFu) |
                  ((tensorRows & 0xFFFFu) << 16));               // dim0 hi | dim1 lo
    g1[3] = (int)(((tensorRows >> 16) & 0xFFFFu) | (32u << 16)); // dim1 hi | tile_dim0
    g1[4] = (int)(tileRows & 0xFFFFu);                           // tile_dim1 (tile_dim2=0)
    g1[5] = (int)Kdim;                                           // tensor_dim0_stride lo
    g1[6] = 0;                                                   // stride0 hi | stride1 lo
    g1[7] = 0;
    v4i z4 = {0, 0, 0, 0};
#if __clang_major__ >= 23
    v8i z8 = {0, 0, 0, 0, 0, 0, 0, 0};
    __builtin_amdgcn_tensor_load_to_lds(g0, g1, z4, z4, z8, 0);
#else
    __builtin_amdgcn_tensor_load_to_lds(g0, g1, z4, z4, 0);
#endif
}
#endif

// ---------------------------------------------------------------- h = x + pos
__global__ __launch_bounds__(256) void hpos_kernel(const float* __restrict__ x,
                                                   const float* __restrict__ pos,
                                                   __bf16* __restrict__ hbf) {
    int row = blockIdx.x;                 // b*S + t
    int t   = row & (SS - 1);
    int pr  = t & (CC - 1);
    int d0  = threadIdx.x * 4;
    float4 xv = *(const float4*)(x + (size_t)row * DD + d0);
    float4 pv = *(const float4*)(pos + (size_t)pr * DD + d0);
    __bf16* o = hbf + (size_t)row * DD + d0;
    o[0] = f2bf(xv.x + pv.x); o[1] = f2bf(xv.y + pv.y);
    o[2] = f2bf(xv.z + pv.z); o[3] = f2bf(xv.w + pv.w);
}

// ------------------------------------------- weight convert+transpose fp32->bf16
// WT[n*K + k] = bf16(W[k*N + n])
__global__ __launch_bounds__(256) void wconvert_kernel(const float* __restrict__ W,
                                                       __bf16* __restrict__ WT,
                                                       int K, int N) {
    int idx = blockIdx.x * 256 + threadIdx.x;   // idx = n*K + k
    int k = idx % K;
    int n = idx / K;
    WT[idx] = f2bf(W[(size_t)k * N + n]);
}

// ---------------------------------------------------------------- WMMA GEMM
// C(M,N) = A(M,K) * B(K,N), A row-major bf16, B given as BT[n*K+k] bf16.
// Block tile 128x128, BK=32, 256 threads = 8 wave32 waves, wave tile 32x64.
// Tiles staged by the Tensor Data Mover (waves 0/1 own A/B descriptors),
// double-buffered, synchronized with TENSORcnt + workgroup barrier.
// EPI: 0 = store bf16           1 = store bf16 * scale
//      2 = store f32            3 = relu(acc+bias) -> bf16
//      4 = acc+bias -> f32
template <int EPI>
__global__ __launch_bounds__(256) void gemm_bf16(const __bf16* __restrict__ A,
                                                 const __bf16* __restrict__ BT,
                                                 const float* __restrict__ bias,
                                                 void* __restrict__ Cout,
                                                 int Mdim, int Ndim, int Kdim,
                                                 float scale) {
    __shared__ __align__(16) __bf16 As[2][128 * 40];  // stride 40 (80B, 16B-aligned)
    __shared__ __align__(16) __bf16 Bs[2][128 * 40];

    const int tid  = threadIdx.x;
    const int lane = tid & 31;
    const int wid  = tid >> 5;           // wave32
    const int half = lane >> 4;
    const int l16  = lane & 15;
    const int wm   = wid >> 1;           // 0..3
    const int wn   = wid & 1;            // 0..1
    const int rowBase = blockIdx.y * 128;
    const int colBase = blockIdx.x * 128;
    const int nk   = Kdim >> 5;

    v8f acc[2][4];
    v8f zero = {0.f, 0.f, 0.f, 0.f, 0.f, 0.f, 0.f, 0.f};
#pragma unroll
    for (int i = 0; i < 2; ++i)
#pragma unroll
        for (int j = 0; j < 4; ++j) acc[i][j] = zero;

#if USE_TDM
    // prologue: issue DMA for k-step 0 (wave0 -> A tile, wave1 -> B tile)
    if (wid < 2) {
        const __bf16* g = (wid == 0) ? (A  + (size_t)rowBase * Kdim)
                                     : (BT + (size_t)colBase * Kdim);
        u32 lds = (u32)(__SIZE_TYPE__)((wid == 0) ? &As[0][0] : &Bs[0][0]);
        tdm_load_tile(lds, g, (u32)Kdim, (u32)Mdim, 128u);
    }
    for (int kt = 0; kt < nk; ++kt) {
        if (wid < 2) __builtin_amdgcn_s_wait_tensorcnt(0);
        __syncthreads();                      // buffer kt&1 is ready for all waves
        if (wid < 2 && (kt + 1) < nk) {       // prefetch next k-step into other buffer
            int k1 = (kt + 1) << 5;
            const __bf16* g = (wid == 0) ? (A  + (size_t)rowBase * Kdim + k1)
                                         : (BT + (size_t)colBase * Kdim + k1);
            u32 lds = (u32)(__SIZE_TYPE__)((wid == 0) ? &As[(kt + 1) & 1][0]
                                                      : &Bs[(kt + 1) & 1][0]);
            tdm_load_tile(lds, g, (u32)Kdim, (u32)Mdim, 128u);
        }
        const __bf16* Ab = &As[kt & 1][0];
        const __bf16* Bb = &Bs[kt & 1][0];
        v16bf af[2];
#pragma unroll
        for (int mt = 0; mt < 2; ++mt)
            af[mt] = frag_a_ld(&Ab[(wm * 32 + mt * 16 + l16) * 40], half);
#pragma unroll
        for (int nt = 0; nt < 4; ++nt) {
            v16bf bf_ = frag_b_ld(&Bb[(wn * 64 + nt * 16 + l16) * 40], half);
#pragma unroll
            for (int mt = 0; mt < 2; ++mt)
                acc[mt][nt] = wmma_bf16(af[mt], bf_, acc[mt][nt]);
        }
    }
#else
    const int sr  = tid >> 1;            // 0..127
    const int seg = tid & 1;             // half-row (16 elems)
    for (int kt = 0; kt < nk; ++kt) {
        int k0 = kt << 5;
        {   // stage A tile (row-major)
            const uint4* src = (const uint4*)(A + (size_t)(rowBase + sr) * Kdim + k0 + seg * 16);
            uint4* dst = (uint4*)(&As[kt & 1][sr * 40 + seg * 16]);
            dst[0] = src[0]; dst[1] = src[1];
        }
        {   // stage B tile ([N][K] rows)
            const uint4* src = (const uint4*)(BT + (size_t)(colBase + sr) * Kdim + k0 + seg * 16);
            uint4* dst = (uint4*)(&Bs[kt & 1][sr * 40 + seg * 16]);
            dst[0] = src[0]; dst[1] = src[1];
        }
        __syncthreads();
        const __bf16* Ab = &As[kt & 1][0];
        const __bf16* Bb = &Bs[kt & 1][0];
        v16bf af[2];
#pragma unroll
        for (int mt = 0; mt < 2; ++mt)
            af[mt] = frag_a_ld(&Ab[(wm * 32 + mt * 16 + l16) * 40], half);
#pragma unroll
        for (int nt = 0; nt < 4; ++nt) {
            v16bf bf_ = frag_b_ld(&Bb[(wn * 64 + nt * 16 + l16) * 40], half);
#pragma unroll
            for (int mt = 0; mt < 2; ++mt)
                acc[mt][nt] = wmma_bf16(af[mt], bf_, acc[mt][nt]);
        }
        __syncthreads();
    }
#endif

    // epilogue: element r of v8f -> (m = r + 8*half, n = l16)
#pragma unroll
    for (int mt = 0; mt < 2; ++mt) {
#pragma unroll
        for (int nt = 0; nt < 4; ++nt) {
            int gr0 = rowBase + wm * 32 + mt * 16 + half * 8;
            int gc  = colBase + wn * 64 + nt * 16 + l16;
#pragma unroll
            for (int r = 0; r < 8; ++r) {
                float v = acc[mt][nt][r];
                size_t off = (size_t)(gr0 + r) * Ndim + gc;
                if (EPI == 0) {
                    ((__bf16*)Cout)[off] = f2bf(v);
                } else if (EPI == 1) {
                    ((__bf16*)Cout)[off] = f2bf(v * scale);
                } else if (EPI == 2) {
                    ((float*)Cout)[off] = v;
                } else if (EPI == 3) {
                    ((__bf16*)Cout)[off] = f2bf(fmaxf(v + bias[gc], 0.f));
                } else {
                    ((float*)Cout)[off] = v + bias[gc];
                }
            }
        }
    }
}

// ------------------------------------------------- sliding-window attention
// grid (S/64, H, B), 256 threads. 64 queries x 128-key window, all WMMA bf16.
__global__ __launch_bounds__(256) void attn_kernel(const __bf16* __restrict__ q,
                                                   const __bf16* __restrict__ k,
                                                   const __bf16* __restrict__ v,
                                                   const float* __restrict__ rpe,
                                                   __bf16* __restrict__ ctx) {
    __shared__ __align__(16) __bf16 Qs[64 * 72];
    __shared__ __align__(16) __bf16 Ks[128 * 72];
    __shared__ __align__(16) __bf16 Vt[64 * 136];    // [dh][key]
    __shared__ __align__(16) float  Sraw[64 * 132];
    __shared__ __align__(16) __bf16 Ps[64 * 136];
    __shared__ float redmax[64 * 4];
    __shared__ float redsum[64 * 4];
    __shared__ float rpes[CC];

    const int t0   = blockIdx.x * 64;
    const int head = blockIdx.y;
    const int b    = blockIdx.z;
    const int tid  = threadIdx.x;
    const int lane = tid & 31;
    const int wid  = tid >> 5;
    const int half = lane >> 4;
    const int l16  = lane & 15;

    const size_t hoff = (size_t)head * DHH;

    // ---- stage Q (64x64)
    {
        int r = tid >> 2, sgq = tid & 3;
        const uint4* src = (const uint4*)(q + ((size_t)b * SS + t0 + r) * DD + hoff + sgq * 16);
        uint4* dst = (uint4*)(&Qs[r * 72 + sgq * 16]);
        dst[0] = src[0]; dst[1] = src[1];
    }
    // ---- stage K window (128x64), zero-pad t<0
    {
        int r = tid >> 1, sg = tid & 1;
        int t = t0 - 64 + r;
        uint4* dst = (uint4*)(&Ks[r * 72 + sg * 32]);
        if (t >= 0) {
            const uint4* src = (const uint4*)(k + ((size_t)b * SS + t) * DD + hoff + sg * 32);
            dst[0] = src[0]; dst[1] = src[1]; dst[2] = src[2]; dst[3] = src[3];
        } else {
            uint4 z = {0u, 0u, 0u, 0u};
            dst[0] = z; dst[1] = z; dst[2] = z; dst[3] = z;
        }
    }
    // ---- stage V transposed: Vt[dh][key]
    {
        int key = tid >> 1, dh0 = (tid & 1) * 32;
        int t = t0 - 64 + key;
        if (t >= 0) {
            const __bf16* src = v + ((size_t)b * SS + t) * DD + hoff + dh0;
#pragma unroll
            for (int i = 0; i < 32; ++i) Vt[(dh0 + i) * 136 + key] = src[i];
        } else {
#pragma unroll
            for (int i = 0; i < 32; ++i) Vt[(dh0 + i) * 136 + key] = f2bf(0.f);
        }
    }
    if (tid < CC) rpes[tid] = rpe[head * CC + tid];
    __syncthreads();

    // ---- scores: wave (mt = wid>>1) x (4 n-tiles), K = 64 -> 2 WMMAs/tile
    {
        const int mt = wid >> 1;
        const int ntg = wid & 1;
        v16bf afQ[2];
#pragma unroll
        for (int ks = 0; ks < 2; ++ks)
            afQ[ks] = frag_a_ld(&Qs[(mt * 16 + l16) * 72 + ks * 32], half);
#pragma unroll
        for (int j = 0; j < 4; ++j) {
            const int ntj = ntg * 4 + j;
            v8f acc = {0.f, 0.f, 0.f, 0.f, 0.f, 0.f, 0.f, 0.f};
#pragma unroll
            for (int ks = 0; ks < 2; ++ks) {
                v16bf bf_ = frag_b_ld(&Ks[(ntj * 16 + l16) * 72 + ks * 32], half);
                acc = wmma_bf16(afQ[ks], bf_, acc);
            }
#pragma unroll
            for (int r = 0; r < 8; ++r) {
                int m    = r + half * 8;
                int tq   = t0 + mt * 16 + m;
                int kidx = ntj * 16 + l16;
                int tk   = t0 - 64 + kidx;
                int d    = tq - tk;
                // branchless: clamp index, load bias unconditionally, one select
                int dc = min(max(d, 0), CC - 1);
                float bias_ = rpes[dc];
                bool ok = (tk >= 0) & (d >= 0) & (d < CC) & (bias_ != 0.f);
                float sv = ok ? (acc[r] + bias_) : FMINF;  // q pre-scaled by 1/8
                Sraw[(mt * 16 + m) * 132 + kidx] = sv;
            }
        }
    }
    __syncthreads();

    // ---- softmax over 128 keys: 4 threads / row
    {
        int row = tid >> 2, qq = tid & 3;
        float* rowp = &Sraw[row * 132 + qq * 32];
        float mx = FMINF;
#pragma unroll
        for (int c = 0; c < 32; ++c) mx = fmaxf(mx, rowp[c]);
        redmax[row * 4 + qq] = mx;
        __syncthreads();
        mx = fmaxf(fmaxf(redmax[row * 4 + 0], redmax[row * 4 + 1]),
                   fmaxf(redmax[row * 4 + 2], redmax[row * 4 + 3]));
        float sum = 0.f;
#pragma unroll
        for (int c = 0; c < 32; ++c) {
            float e = __expf(rowp[c] - mx);
            rowp[c] = e;
            sum += e;
        }
        redsum[row * 4 + qq] = sum;
        __syncthreads();
        sum = redsum[row * 4 + 0] + redsum[row * 4 + 1] +
              redsum[row * 4 + 2] + redsum[row * 4 + 3];
        float inv = 1.f / sum;
#pragma unroll
        for (int c = 0; c < 32; ++c)
            Ps[row * 136 + qq * 32 + c] = f2bf(rowp[c] * inv);
    }
    __syncthreads();

    // ---- O = P(64x128) @ V(128x64) via Vt; wave: 1 m-tile x 2 n-tiles
    {
        const int mt = wid >> 1;
        const int ntg = wid & 1;
        v8f accO[2];
        v8f zero = {0.f, 0.f, 0.f, 0.f, 0.f, 0.f, 0.f, 0.f};
        accO[0] = zero; accO[1] = zero;
#pragma unroll
        for (int ks = 0; ks < 4; ++ks) {
            v16bf afP = frag_a_ld(&Ps[(mt * 16 + l16) * 136 + ks * 32], half);
#pragma unroll
            for (int j = 0; j < 2; ++j) {
                int ntj = ntg * 2 + j;
                v16bf bv = frag_b_ld(&Vt[(ntj * 16 + l16) * 136 + ks * 32], half);
                accO[j] = wmma_bf16(afP, bv, accO[j]);
            }
        }
#pragma unroll
        for (int j = 0; j < 2; ++j) {
            int ntj = ntg * 2 + j;
#pragma unroll
            for (int r = 0; r < 8; ++r) {
                int m  = r + half * 8;
                int tq = t0 + mt * 16 + m;
                int dh = ntj * 16 + l16;
                ctx[((size_t)b * SS + tq) * DD + hoff + dh] = f2bf(accO[j][r]);
            }
        }
    }
}

// ------------------------------------------------- residual + LayerNorm
// y = (x + pos) + proj ; LN -> outf (fp32) and outbf (bf16 for FF GEMM)
__global__ __launch_bounds__(256) void ln1_kernel(const float* __restrict__ x,
                                                  const float* __restrict__ pos,
                                                  const float* __restrict__ proj,
                                                  const float* __restrict__ g,
                                                  const float* __restrict__ beta,
                                                  float* __restrict__ outf,
                                                  __bf16* __restrict__ outbf) {
    __shared__ float rs_[256];
    __shared__ float rq_[256];
    int row = blockIdx.x;
    int t   = row & (SS - 1);
    int pr  = t & (CC - 1);
    int tid = threadIdx.x;
    int d0  = tid * 4;
    float4 xv = *(const float4*)(x    + (size_t)row * DD + d0);
    float4 pv = *(const float4*)(pos  + (size_t)pr  * DD + d0);
    float4 jv = *(const float4*)(proj + (size_t)row * DD + d0);
    float y[4] = {xv.x + pv.x + jv.x, xv.y + pv.y + jv.y,
                  xv.z + pv.z + jv.z, xv.w + pv.w + jv.w};
    float s = 0.f, s2 = 0.f;
#pragma unroll
    for (int i = 0; i < 4; ++i) { s += y[i]; s2 += y[i] * y[i]; }
    rs_[tid] = s; rq_[tid] = s2;
    __syncthreads();
    for (int off = 128; off > 0; off >>= 1) {
        if (tid < off) { rs_[tid] += rs_[tid + off]; rq_[tid] += rq_[tid + off]; }
        __syncthreads();
    }
    float mu  = rs_[0] * (1.f / DD);
    float var = rq_[0] * (1.f / DD) - mu * mu;
    float rsd = rsqrtf(var + 1e-5f);
    float4 gv = *(const float4*)(g + d0);
    float4 bv = *(const float4*)(beta + d0);
    float o0 = (y[0] - mu) * rsd * gv.x + bv.x;
    float o1 = (y[1] - mu) * rsd * gv.y + bv.y;
    float o2 = (y[2] - mu) * rsd * gv.z + bv.z;
    float o3 = (y[3] - mu) * rsd * gv.w + bv.w;
    float4* of = (float4*)(outf + (size_t)row * DD + d0);
    *of = make_float4(o0, o1, o2, o3);
    __bf16* ob = outbf + (size_t)row * DD + d0;
    ob[0] = f2bf(o0); ob[1] = f2bf(o1); ob[2] = f2bf(o2); ob[3] = f2bf(o3);
}

// y = ff2 + attn ; LN -> out (fp32, final output)
__global__ __launch_bounds__(256) void ln2_kernel(const float* __restrict__ ff2,
                                                  const float* __restrict__ attn,
                                                  const float* __restrict__ g,
                                                  const float* __restrict__ beta,
                                                  float* __restrict__ out) {
    __shared__ float rs_[256];
    __shared__ float rq_[256];
    int row = blockIdx.x;
    int tid = threadIdx.x;
    int d0  = tid * 4;
    float4 fv = *(const float4*)(ff2  + (size_t)row * DD + d0);
    float4 av = *(const float4*)(attn + (size_t)row * DD + d0);
    float y[4] = {fv.x + av.x, fv.y + av.y, fv.z + av.z, fv.w + av.w};
    float s = 0.f, s2 = 0.f;
#pragma unroll
    for (int i = 0; i < 4; ++i) { s += y[i]; s2 += y[i] * y[i]; }
    rs_[tid] = s; rq_[tid] = s2;
    __syncthreads();
    for (int off = 128; off > 0; off >>= 1) {
        if (tid < off) { rs_[tid] += rs_[tid + off]; rq_[tid] += rq_[tid + off]; }
        __syncthreads();
    }
    float mu  = rs_[0] * (1.f / DD);
    float var = rq_[0] * (1.f / DD) - mu * mu;
    float rsd = rsqrtf(var + 1e-5f);
    float4 gv = *(const float4*)(g + d0);
    float4 bv = *(const float4*)(beta + d0);
    float4* of = (float4*)(out + (size_t)row * DD + d0);
    *of = make_float4((y[0] - mu) * rsd * gv.x + bv.x,
                      (y[1] - mu) * rsd * gv.y + bv.y,
                      (y[2] - mu) * rsd * gv.z + bv.z,
                      (y[3] - mu) * rsd * gv.w + bv.w);
}

// ---------------------------------------------------------------- host launch
extern "C" void kernel_launch(void* const* d_in, const int* in_sizes, int n_in,
                              void* d_out, int out_size, void* d_ws, size_t ws_size,
                              hipStream_t stream) {
    const float* x    = (const float*)d_in[0];
    // d_in[1] = causal_mask (unused; causality handled analytically)
    const float* pos  = (const float*)d_in[2];
    const float* rpe  = (const float*)d_in[3];
    const float* Wq   = (const float*)d_in[4];
    const float* Wk   = (const float*)d_in[5];
    const float* Wv   = (const float*)d_in[6];
    const float* Wo   = (const float*)d_in[7];
    const float* ln1s = (const float*)d_in[8];
    const float* ln1o = (const float*)d_in[9];
    const float* W1   = (const float*)d_in[10];
    const float* b1   = (const float*)d_in[11];
    const float* W2   = (const float*)d_in[12];
    const float* b2   = (const float*)d_in[13];
    const float* ln2s = (const float*)d_in[14];
    const float* ln2o = (const float*)d_in[15];

    size_t off = 0;
    auto alloc = [&](size_t bytes) -> void* {
        void* p = (char*)d_ws + off;
        off += (bytes + 255) & ~(size_t)255;
        return p;
    };
    __bf16* hbf    = (__bf16*)alloc((size_t)MM * DD * 2);
    __bf16* qbf    = (__bf16*)alloc((size_t)MM * DD * 2);
    __bf16* kbf    = (__bf16*)alloc((size_t)MM * DD * 2);
    __bf16* vbf    = (__bf16*)alloc((size_t)MM * DD * 2);
    __bf16* ctxbf  = (__bf16*)alloc((size_t)MM * DD * 2);
    float*  projf  = (float*) alloc((size_t)MM * DD * 4);
    float*  attnf  = (float*) alloc((size_t)MM * DD * 4);
    __bf16* attnbf = (__bf16*)alloc((size_t)MM * DD * 2);
    __bf16* ff1bf  = (__bf16*)alloc((size_t)MM * DFF * 2);
    float*  ff2f   = (float*) alloc((size_t)MM * DD * 4);
    __bf16* WTq    = (__bf16*)alloc((size_t)DD * DD * 2);
    __bf16* WTk    = (__bf16*)alloc((size_t)DD * DD * 2);
    __bf16* WTv    = (__bf16*)alloc((size_t)DD * DD * 2);
    __bf16* WTo    = (__bf16*)alloc((size_t)DD * DD * 2);
    __bf16* WT1    = (__bf16*)alloc((size_t)DD * DFF * 2);
    __bf16* WT2    = (__bf16*)alloc((size_t)DFF * DD * 2);

    // 1) h = x + pos  (bf16)
    hpos_kernel<<<MM, 256, 0, stream>>>(x, pos, hbf);

    // 2) weight convert + transpose
    wconvert_kernel<<<(DD * DD) / 256, 256, 0, stream>>>(Wq, WTq, DD, DD);
    wconvert_kernel<<<(DD * DD) / 256, 256, 0, stream>>>(Wk, WTk, DD, DD);
    wconvert_kernel<<<(DD * DD) / 256, 256, 0, stream>>>(Wv, WTv, DD, DD);
    wconvert_kernel<<<(DD * DD) / 256, 256, 0, stream>>>(Wo, WTo, DD, DD);
    wconvert_kernel<<<(DD * DFF) / 256, 256, 0, stream>>>(W1, WT1, DD, DFF);
    wconvert_kernel<<<(DFF * DD) / 256, 256, 0, stream>>>(W2, WT2, DFF, DD);

    // 3) QKV projections (q pre-scaled by 1/sqrt(DH) = 0.125)
    dim3 gproj(DD / 128, MM / 128);
    gemm_bf16<1><<<gproj, 256, 0, stream>>>(hbf, WTq, nullptr, qbf, MM, DD, DD, 0.125f);
    gemm_bf16<0><<<gproj, 256, 0, stream>>>(hbf, WTk, nullptr, kbf, MM, DD, DD, 1.f);
    gemm_bf16<0><<<gproj, 256, 0, stream>>>(hbf, WTv, nullptr, vbf, MM, DD, DD, 1.f);

    // 4) sliding-window attention
    attn_kernel<<<dim3(SS / 64, HH, BB), 256, 0, stream>>>(qbf, kbf, vbf, rpe, ctxbf);

    // 5) output projection (fp32 result for residual+LN)
    gemm_bf16<2><<<gproj, 256, 0, stream>>>(ctxbf, WTo, nullptr, projf, MM, DD, DD, 1.f);

    // 6) residual + LN1
    ln1_kernel<<<MM, 256, 0, stream>>>(x, pos, projf, ln1s, ln1o, attnf, attnbf);

    // 7) FF1: relu(attn @ W1 + b1) -> bf16
    gemm_bf16<3><<<dim3(DFF / 128, MM / 128), 256, 0, stream>>>(attnbf, WT1, b1, ff1bf,
                                                                MM, DFF, DD, 1.f);
    // 8) FF2: ff1 @ W2 + b2 -> fp32
    gemm_bf16<4><<<gproj, 256, 0, stream>>>(ff1bf, WT2, b2, ff2f, MM, DD, DFF, 1.f);

    // 9) residual + LN2 -> output
    ln2_kernel<<<MM, 256, 0, stream>>>(ff2f, attnf, ln2s, ln2o, (float*)d_out);
}
